// ProteinACmix_86114094285218
// MI455X (gfx1250) — matrix operand order repaired and checked by reading, not compiled, not run on hardware
//
#include <hip/hip_runtime.h>

#define LSEQ 2048
#define BATCH 32
#define CIN 128
#define MTOT 384   // q,k,v packed
#define NHEAD 8
#define HD 16

typedef __attribute__((ext_vector_type(16))) _Float16 v16h;
typedef __attribute__((ext_vector_type(8)))  _Float16 v8h;
typedef __attribute__((ext_vector_type(8)))  float    v8f;

// ---------------- embedding gather: x_h[b,l,c] = (f16) emb[tok[b,l], c] ----
__global__ __launch_bounds__(256)
void embed_kernel(const int* __restrict__ tokens, const float* __restrict__ emb,
                  _Float16* __restrict__ xh)
{
  int i = blockIdx.x * blockDim.x + threadIdx.x;   // over B*L
  if (i >= BATCH * LSEQ) return;
  int t = tokens[i];
  const float4* src = (const float4*)(emb + (long)t * CIN);
  _Float16* dst = xh + (long)i * CIN;
#pragma unroll
  for (int j = 0; j < CIN / 4; ++j) {
    float4 f = src[j];
    dst[4 * j + 0] = (_Float16)f.x;
    dst[4 * j + 1] = (_Float16)f.y;
    dst[4 * j + 2] = (_Float16)f.z;
    dst[4 * j + 3] = (_Float16)f.w;
  }
}

// ---------------- pack Wq|Wk|Wv -> f16 [384,128], bias -> f32 [384] --------
__global__ __launch_bounds__(256)
void pack_w_kernel(const float* __restrict__ wq, const float* __restrict__ wk,
                   const float* __restrict__ wv, const float* __restrict__ bq,
                   const float* __restrict__ bk, const float* __restrict__ bv,
                   _Float16* __restrict__ Wh, float* __restrict__ bias)
{
  int i = blockIdx.x * blockDim.x + threadIdx.x;   // over 384*128
  if (i < MTOT * CIN) {
    int m = i / CIN, c = i % CIN;
    const float* w = (m < CIN) ? wq : (m < 2 * CIN) ? wk : wv;
    Wh[i] = (_Float16)w[(m % CIN) * CIN + c];
  }
  if (i < MTOT)
    bias[i] = (i < CIN) ? bq[i] : (i < 2 * CIN) ? bk[i - CIN] : bv[i - 2 * CIN];
}

// ---------------- positional encoding pe[d,l] ------------------------------
__global__ __launch_bounds__(256)
void pe_kernel(const float* __restrict__ wp, const float* __restrict__ bp,
               float* __restrict__ pe)
{
  int i = blockIdx.x * blockDim.x + threadIdx.x;   // over 16*L
  if (i >= HD * LSEQ) return;
  int d = i / LSEQ, l = i % LSEQ;
  float loc = -1.0f + 2.0f * (float)l / (float)(LSEQ - 1);
  pe[i] = wp[d * 2 + 0] * loc - wp[d * 2 + 1] + bp[d];
}

// ---------------- QKV GEMM via WMMA f32_16x16x32_f16 -----------------------
// X: [B, L, CIN] f16 ; W: [MTOT, CIN] f16 ; out: [B, MTOT, L] f32
// wave tile: 32(M) x 64(N); K=128 in 4 steps. 32 WMMAs / wave, EXEC all-1s.
// Per K-step: 8 WMMAs vs 12x16B loads (each A and B fragment feeds multiple
// WMMAs), which doubles arithmetic intensity vs a 16x64 tile.
__global__ __launch_bounds__(256)
void qkv_gemm_kernel(const _Float16* __restrict__ X, const _Float16* __restrict__ W,
                     const float* __restrict__ bias, float* __restrict__ out)
{
  const int wave = threadIdx.x >> 5;
  const int lane = threadIdx.x & 31;
  const int gw   = blockIdx.x * 8 + wave;       // B * 12 * 32 waves total
  const int b    = gw / (12 * 32);
  const int rem  = gw % (12 * 32);
  const int m0   = (rem / 32) * 32;             // 32-row M supertile
  const int n0   = (rem % 32) * 64;             // 64-col N supertile
  const int g    = lane >> 4;                   // lane half
  const int r    = lane & 15;                   // row within A / col within B

  v8f acc[2][4];
#pragma unroll
  for (int mt = 0; mt < 2; ++mt)
#pragma unroll
    for (int j = 0; j < 4; ++j)
      acc[mt][j] = (v8f){0.f,0.f,0.f,0.f,0.f,0.f,0.f,0.f};

#pragma unroll
  for (int ks = 0; ks < 4; ++ks) {
    const int k0 = ks * 32;
    // A fragments: rows m0+16*mt+r, runs K=[8g..8g+7] and [16+8g..16+8g+7]
    union { v16h v; v8h h[2]; } ua[2];
#pragma unroll
    for (int mt = 0; mt < 2; ++mt) {
      const _Float16* wrow = W + (long)(m0 + 16 * mt + r) * CIN + k0;
      ua[mt].h[0] = *(const v8h*)(wrow + 8 * g);
      ua[mt].h[1] = *(const v8h*)(wrow + 16 + 8 * g);
    }
#pragma unroll
    for (int j = 0; j < 4; ++j) {
      const int l = n0 + 16 * j + r;            // B column
      const _Float16* xp = X + ((long)b * LSEQ + l) * CIN + k0 + 8 * g;
      union { v16h v; v8h h[2]; } ub;
      ub.h[0] = *(const v8h*)(xp);
      ub.h[1] = *(const v8h*)(xp + 16);
#pragma unroll
      for (int mt = 0; mt < 2; ++mt)
        acc[mt][j] = __builtin_amdgcn_wmma_f32_16x16x32_f16(
            false, ua[mt].v, false, ub.v, (short)0, acc[mt][j], false, false);
    }
  }

#pragma unroll
  for (int mt = 0; mt < 2; ++mt)
#pragma unroll
    for (int j = 0; j < 4; ++j) {
      const int l = n0 + 16 * j + r;
#pragma unroll
      for (int rr = 0; rr < 8; ++rr) {
        const int m = m0 + 16 * mt + rr + 8 * g;  // D: VGPR rr -> M=rr+8g, N=r
        out[((long)b * MTOT + m) * LSEQ + l] = acc[mt][j][rr] + bias[m];
      }
    }
}

// ---------------- f_all[b,o,d,l] = sum_w wf[o,w]*qkv[b, w*16+d, l] ---------
__global__ __launch_bounds__(256)
void fall_kernel(const float* __restrict__ qkv, const float* __restrict__ wf,
                 float* __restrict__ fall)
{
  long i = (long)blockIdx.x * blockDim.x + threadIdx.x;  // B*9*16*L
  if (i >= (long)BATCH * 9 * HD * LSEQ) return;
  int l = (int)(i % LSEQ); long t = i / LSEQ;
  int d = (int)(t % HD);   t /= HD;
  int o = (int)(t % 9);
  int b = (int)(t / 9);
  const float* base = qkv + ((long)b * MTOT + d) * LSEQ + l;   // stride 16*L over w
  float s = 0.f;
#pragma unroll
  for (int w = 0; w < 24; ++w) s += wf[o * 24 + w] * base[(long)w * HD * LSEQ];
  fall[i] = s;
}

// ---------------- attention(window7) + depthwise conv + combine + relu -----
// one thread per (b, h, l) -> 16 channels c = h*16+d
__global__ __launch_bounds__(256)
void attconv_kernel(const float* __restrict__ qkv, const float* __restrict__ pe,
                    const float* __restrict__ fall, const float* __restrict__ wd,
                    const float* __restrict__ bd, const float* __restrict__ r1p,
                    const float* __restrict__ r2p, float* __restrict__ y)
{
  long i = (long)blockIdx.x * blockDim.x + threadIdx.x;  // B*8*L
  if (i >= (long)BATCH * NHEAD * LSEQ) return;
  int l = (int)(i % LSEQ); long t = i / LSEQ;
  int h = (int)(t % NHEAD);
  int b = (int)(t / NHEAD);

  const float* q = qkv + ((long)b * MTOT + h * HD) * LSEQ;
  const float* k = qkv + ((long)b * MTOT + CIN + h * HD) * LSEQ;
  const float* v = qkv + ((long)b * MTOT + 2 * CIN + h * HD) * LSEQ;

  int jr[7];
#pragma unroll
  for (int w = 0; w < 7; ++w) {
    int j = l + w - 3;
    jr[w] = (j < 0) ? -j : ((j >= LSEQ) ? 2 * LSEQ - 2 - j : j);
  }

  float qv[16];
#pragma unroll
  for (int d = 0; d < HD; ++d) qv[d] = q[(long)d * LSEQ + l] * 0.25f;  // hd^-0.5

  float att[7], mx = -1e30f;
#pragma unroll
  for (int w = 0; w < 7; ++w) {
    int j = jr[w];
    float s = 0.f;
#pragma unroll
    for (int d = 0; d < HD; ++d) {
      float term = k[(long)d * LSEQ + j] + pe[d * LSEQ + l] - pe[d * LSEQ + j];
      s += qv[d] * term;
    }
    att[w] = s;
    mx = fmaxf(mx, s);
  }
  float den = 0.f;
#pragma unroll
  for (int w = 0; w < 7; ++w) { att[w] = expf(att[w] - mx); den += att[w]; }
  const float inv = 1.f / den;
  const float r1 = r1p[0], r2 = r2p[0];

#pragma unroll
  for (int d = 0; d < HD; ++d) {
    float oa = 0.f;
#pragma unroll
    for (int w = 0; w < 7; ++w) oa += att[w] * v[(long)d * LSEQ + jr[w]];
    oa *= inv;

    const int c  = h * HD + d;
    const int gg = c >> 3;                      // depthwise group = c/8
    float oc = bd[c];
#pragma unroll
    for (int o = 0; o < 9; ++o) {
      const float* fb = fall + (((long)b * 9 + o) * HD + gg) * LSEQ;
#pragma unroll
      for (int tt = 0; tt < 3; ++tt) {
        int jj = l + tt - 1;
        float fv = (jj < 0 || jj >= LSEQ) ? 0.f : fb[jj];
        oc += wd[(c * 9 + o) * 3 + tt] * fv;
      }
    }
    y[((long)b * CIN + c) * LSEQ + l] = fmaxf(r1 * oa + r2 * oc, 0.f);
  }
}

// ---------------- batchnorm: stats then apply ------------------------------
__global__ void zero_kernel(float* __restrict__ p, int n)
{
  int i = blockIdx.x * blockDim.x + threadIdx.x;
  if (i < n) p[i] = 0.f;
}

__global__ __launch_bounds__(256)
void bnstats_kernel(const float* __restrict__ y, float* __restrict__ stats)
{
  const int c = blockIdx.y;
  long idx = (long)blockIdx.x * blockDim.x + threadIdx.x;  // over B*L
  float val = 0.f;
  if (idx < (long)BATCH * LSEQ) {
    long b = idx / LSEQ, l = idx % LSEQ;
    val = y[(b * CIN + c) * LSEQ + l];
  }
  __shared__ float s1[256], s2[256];
  s1[threadIdx.x] = val;
  s2[threadIdx.x] = val * val;
  __syncthreads();
  for (int s = 128; s > 0; s >>= 1) {
    if ((int)threadIdx.x < s) {
      s1[threadIdx.x] += s1[threadIdx.x + s];
      s2[threadIdx.x] += s2[threadIdx.x + s];
    }
    __syncthreads();
  }
  if (threadIdx.x == 0) {
    atomicAdd(&stats[c], s1[0]);
    atomicAdd(&stats[CIN + c], s2[0]);
  }
}

__global__ __launch_bounds__(256)
void bnapply_kernel(const float* __restrict__ y, const float* __restrict__ stats,
                    const float* __restrict__ gamma, const float* __restrict__ beta,
                    _Float16* __restrict__ xnext, float* __restrict__ outf,
                    int writeNext, int writeOut)
{
  long i = (long)blockIdx.x * blockDim.x + threadIdx.x;  // B*128*L
  if (i >= (long)BATCH * CIN * LSEQ) return;
  int l = (int)(i % LSEQ); long t = i / LSEQ;
  int c = (int)(t % CIN);
  int b = (int)(t / CIN);
  const float n = (float)(BATCH * LSEQ);
  float m  = stats[c] / n;
  float vr = stats[CIN + c] / n - m * m;
  float sc = rsqrtf(vr + 1e-5f);
  float val = (y[i] - m) * sc * gamma[c] + beta[c];
  if (writeNext) xnext[((long)b * LSEQ + l) * CIN + c] = (_Float16)val;
  if (writeOut)  outf[i] = val;
}

// ---------------------------------------------------------------------------
extern "C" void kernel_launch(void* const* d_in, const int* in_sizes, int n_in,
                              void* d_out, int out_size, void* d_ws, size_t ws_size,
                              hipStream_t stream)
{
  (void)in_sizes; (void)n_in; (void)out_size; (void)ws_size;

  const int*   tokens = (const int*)d_in[0];
  const float* emb    = (const float*)d_in[1];
  // per-layer param index bases: layer1 = 2, layer2 = 17 (15 params each)
  const int base[2] = {2, 17};

  // workspace bump allocator
  char* p = (char*)d_ws;
  auto alloc = [&](size_t bytes) {
    char* r = p;
    p += (bytes + 255) & ~(size_t)255;
    return (void*)r;
  };
  _Float16* xh    = (_Float16*)alloc((size_t)BATCH * LSEQ * CIN * 2);     // f16 act [B,L,C]
  _Float16* Wh    = (_Float16*)alloc((size_t)MTOT * CIN * 2);             // packed weights
  float*    biasq = (float*)alloc((size_t)MTOT * 4);
  float*    qkv   = (float*)alloc((size_t)BATCH * MTOT * LSEQ * 4);       // [B,384,L]
  float*    pe    = (float*)alloc((size_t)HD * LSEQ * 4);
  float*    fall  = (float*)alloc((size_t)BATCH * 9 * HD * LSEQ * 4);     // [B,9,16,L]
  float*    y     = (float*)alloc((size_t)BATCH * CIN * LSEQ * 4);        // [B,128,L]
  float*    stats = (float*)alloc(2 * CIN * 4);

  // embedding gather
  embed_kernel<<<(BATCH * LSEQ + 255) / 256, 256, 0, stream>>>(tokens, emb, xh);

  for (int layer = 0; layer < 2; ++layer) {
    const int bi = base[layer];
    const float* wq = (const float*)d_in[bi + 0];
    const float* bq = (const float*)d_in[bi + 1];
    const float* wk = (const float*)d_in[bi + 2];
    const float* bk = (const float*)d_in[bi + 3];
    const float* wv = (const float*)d_in[bi + 4];
    const float* bv = (const float*)d_in[bi + 5];
    const float* wp = (const float*)d_in[bi + 6];
    const float* bp = (const float*)d_in[bi + 7];
    const float* wf = (const float*)d_in[bi + 8];
    const float* wd = (const float*)d_in[bi + 9];
    const float* bd = (const float*)d_in[bi + 10];
    const float* ra = (const float*)d_in[bi + 11];
    const float* rb = (const float*)d_in[bi + 12];
    const float* gm = (const float*)d_in[bi + 13];
    const float* bt = (const float*)d_in[bi + 14];

    pack_w_kernel<<<(MTOT * CIN + 255) / 256, 256, 0, stream>>>(wq, wk, wv, bq, bk, bv, Wh, biasq);

    // WMMA GEMM: B * 12 * 32 waves (32M x 64N tiles), 8 waves per block
    qkv_gemm_kernel<<<(BATCH * 12 * 32) / 8, 256, 0, stream>>>(xh, Wh, biasq, qkv);

    pe_kernel<<<(HD * LSEQ + 255) / 256, 256, 0, stream>>>(wp, bp, pe);

    fall_kernel<<<(int)(((long)BATCH * 9 * HD * LSEQ + 255) / 256), 256, 0, stream>>>(qkv, wf, fall);

    attconv_kernel<<<(int)(((long)BATCH * NHEAD * LSEQ + 255) / 256), 256, 0, stream>>>(
        qkv, pe, fall, wd, bd, ra, rb, y);

    zero_kernel<<<1, 256, 0, stream>>>(stats, 2 * CIN);

    dim3 sg((BATCH * LSEQ + 255) / 256, CIN);
    bnstats_kernel<<<sg, 256, 0, stream>>>(y, stats);

    bnapply_kernel<<<(int)(((long)BATCH * CIN * LSEQ + 255) / 256), 256, 0, stream>>>(
        y, stats, gm, bt, xh, (float*)d_out, /*writeNext=*/(layer == 0), /*writeOut=*/(layer == 1));
  }
}
